// NODEModel_75548474736884
// MI455X (gfx1250) — matrix operand order repaired and checked by reading, not compiled, not run on hardware
//
#include <hip/hip_runtime.h>

typedef __attribute__((ext_vector_type(16))) _Float16 v16h;
typedef __attribute__((ext_vector_type(8)))  _Float16 v8h;
typedef __attribute__((ext_vector_type(8)))  float    v8f;

namespace {

constexpr int NSTACK = 3;     // hidden stacks (DEPTH-1)
constexpr int SW = 136;       // LDS row stride in halves (272B: 16B aligned, bank-spread)
constexpr int SH = 136;
constexpr float EPSF = 1e-8f;
constexpr float LOG2E_X2 = 2.8853900817779268f;  // 2*log2(e)

// Single-instruction HW transcendentals
__device__ __forceinline__ float rcp_hw(float x)  { return __builtin_amdgcn_rcpf(x); }
__device__ __forceinline__ float sqrt_hw(float x) { return __builtin_amdgcn_sqrtf(x); }
__device__ __forceinline__ float exp2_hw(float x) { return __builtin_amdgcn_exp2f(x); }

// tanh: native V_TANH_F32 (gfx1250 TRANS op, co-executes with WMMA) when the
// toolchain exposes it; otherwise branch-free 1 - 2/(exp(2x)+1).
__device__ __forceinline__ float tanh_fast(float x) {
#if __has_builtin(__builtin_amdgcn_tanhf)
  return __builtin_amdgcn_tanhf(x);
#else
  float e = exp2_hw(x * LOG2E_X2);
  return 1.0f - 2.0f * rcp_hw(e + 1.0f);
#endif
}

__device__ __forceinline__ void sph_features(const float* __restrict__ tx4, float* f) {
  float x = tx4[1], y = tx4[2], z = tx4[3];
  float r   = sqrt_hw(x * x + y * y + z * z + EPSF);
  float rho = sqrt_hw(x * x + y * y + EPSF);
  float ir   = rcp_hw(r);
  float irho = rcp_hw(rho);
  f[0] = r * rcp_hw(1.0f + r);
  f[1] = z * ir;
  f[2] = rho * ir;
  f[3] = x * irho;
  f[4] = y * irho;
}

__device__ __forceinline__ v8f wmma_f16(v16h a, v16h b, v8f c) {
  return __builtin_amdgcn_wmma_f32_16x16x32_f16(false, a, false, b, (short)0, c,
                                                false, false);
}

// A(16x32 f16) fragment: lane holds two contiguous 8-half chunks (K = g*8.. and 16+g*8..)
__device__ __forceinline__ v16h ld_afrag(const _Float16* p) {
  v16h a;
  ((v8h*)&a)[0] = *(const v8h*)p;
  ((v8h*)&a)[1] = *(const v8h*)(p + 16);
  return a;
}
// B(32x16 f16) fragment from transposed weights: 16 contiguous halves (K = g*16..g*16+15)
__device__ __forceinline__ v16h ld_bfrag(const _Float16* p) {
  v16h b;
  ((v8h*)&b)[0] = *(const v8h*)p;
  ((v8h*)&b)[1] = *(const v8h*)(p + 8);
  return b;
}

// Load 128x128 f32 weight W[k][n] transposed into sWT[n*SW + k] as f16 (coalesced reads)
__device__ __forceinline__ void load_wt128(_Float16* sWT, const float* __restrict__ W,
                                           int tid) {
  for (int idx = tid; idx < 128 * 128; idx += 256) {
    int k = idx >> 7, n = idx & 127;
    sWT[n * SW + k] = (_Float16)W[idx];   // W[idx] == W[k*128 + n]
  }
}

// Load F x 128 layer-0 weight, transposed + zero padded in K up to 32
template<int F>
__device__ __forceinline__ void load_wt0(_Float16* sWT, const float* __restrict__ W0,
                                         int tid) {
  for (int idx = tid; idx < 128 * 32; idx += 256) {
    int k = idx >> 7, n = idx & 127;
    sWT[n * SW + k] = (k < F) ? (_Float16)W0[k * 128 + n] : (_Float16)0.0f;
  }
}

// Hidden layer GEMM: acc[tile] = H(MROWSx128) @ W(128x128), bias captured to regs
template<int MROWS>
__device__ __forceinline__ void gemm_hidden(const _Float16* sH, const _Float16* sWT,
                                            const float* sB, int tid,
                                            v8f* acc, float* biasr) {
  constexpr int NT  = 8;
  constexpr int TPW = (MROWS / 16) * NT / 8;   // tiles per wave (8 waves)
  const int wave = tid >> 5, lane = tid & 31, lm = lane & 15, lg = lane >> 4;
  #pragma unroll
  for (int i = 0; i < TPW; ++i) {
    const int tile = wave * TPW + i;
    const int mt = tile / NT, nt = tile % NT;
    v8f c;
    #pragma unroll
    for (int v = 0; v < 8; ++v) c[v] = 0.0f;
    #pragma unroll
    for (int kt = 0; kt < 4; ++kt) {
      v16h a = ld_bfrag(sH + (mt * 16 + lm) * SH + kt * 32 + lg * 16);
      v16h b = ld_bfrag(sWT + (nt * 16 + lm) * SW + kt * 32 + lg * 16);
      c = wmma_f16(a, b, c);
    }
    acc[i]   = c;
    biasr[i] = sB[nt * 16 + lm];
  }
}

// tanh(acc + bias) -> sH (in place; caller barriers around this)
template<int MROWS>
__device__ __forceinline__ void store_act(_Float16* sH, int tid,
                                          const v8f* acc, const float* biasr) {
  constexpr int NT  = 8;
  constexpr int TPW = (MROWS / 16) * NT / 8;
  const int wave = tid >> 5, lane = tid & 31, lm = lane & 15, lg = lane >> 4;
  #pragma unroll
  for (int i = 0; i < TPW; ++i) {
    const int tile = wave * TPW + i;
    const int mt = tile / NT, nt = tile % NT;
    #pragma unroll
    for (int v = 0; v < 8; ++v) {
      float val = tanh_fast(acc[i][v] + biasr[i]);
      sH[(mt * 16 + lg * 8 + v) * SH + nt * 16 + lm] = (_Float16)val;
    }
  }
}

} // namespace

// ---------------- IC MLP: phi0 over N points (64 points / block) ----------------
__global__ __launch_bounds__(256) void ic_mlp_kernel(
    const float* __restrict__ tx, const float* __restrict__ W0,
    const float* __restrict__ b0, const float* __restrict__ Wh,
    const float* __restrict__ bh, const float* __restrict__ Wo,
    const float* __restrict__ bo, float* __restrict__ phi0) {
  constexpr int P = 64, MROWS = 64, NT = 8, TPW = 4, F = 5;
  __shared__ _Float16 sWT[128 * SW];     // 34816 B
  __shared__ _Float16 sH[MROWS * SH];    // 17408 B
  __shared__ float sB[128];
  __shared__ float sWo[128];
  const int tid = threadIdx.x;
  const int wave = tid >> 5, lane = tid & 31, lm = lane & 15, lg = lane >> 4;
  const int base = blockIdx.x * P;

  load_wt0<F>(sWT, W0, tid);
  if (tid < 128) { sB[tid] = b0[tid]; sWo[tid] = Wo[tid]; }
  __syncthreads();

  v8f   acc[TPW];
  float biasr[TPW];

  // ---- layer 0: A fragment built from spherical features in registers ----
  {
    const int mt0 = (wave * TPW) / NT;   // all TPW tiles of this wave share one mt
    v16h afrag;
    #pragma unroll
    for (int h = 0; h < 16; ++h) afrag[h] = (_Float16)0.0f;
    if (lg == 0) {                       // lanes 0-15 hold K=0..7 (covers F=5)
      float f[F];
      sph_features(tx + 4 * (base + mt0 * 16 + lm), f);
      #pragma unroll
      for (int k = 0; k < F; ++k) afrag[k] = (_Float16)f[k];
    }
    #pragma unroll
    for (int i = 0; i < TPW; ++i) {
      const int nt = (wave * TPW + i) % NT;
      v16h b = ld_bfrag(sWT + (nt * 16 + lm) * SW + lg * 16);
      v8f c;
      #pragma unroll
      for (int v = 0; v < 8; ++v) c[v] = 0.0f;
      acc[i]   = wmma_f16(afrag, b, c);
      biasr[i] = sB[nt * 16 + lm];
    }
  }
  __syncthreads();
  store_act<MROWS>(sH, tid, acc, biasr);
  load_wt128(sWT, Wh, tid);
  if (tid < 128) sB[tid] = bh[tid];
  __syncthreads();

  // ---- hidden layers ----
  #pragma unroll 1
  for (int l = 0; l < NSTACK; ++l) {
    gemm_hidden<MROWS>(sH, sWT, sB, tid, acc, biasr);
    __syncthreads();
    store_act<MROWS>(sH, tid, acc, biasr);
    if (l + 1 < NSTACK) {
      load_wt128(sWT, Wh + (l + 1) * 128 * 128, tid);
      if (tid < 128) sB[tid] = bh[(l + 1) * 128 + tid];
    }
    __syncthreads();
  }

  // ---- output layer (128 -> 1) ----
  if (tid < MROWS) {
    float a = bo[0];
    #pragma unroll 8
    for (int k = 0; k < 128; ++k) a += (float)sH[tid * SH + k] * sWo[k];
    phi0[base + tid] = a;
  }
}

// -------- DP MLP: 3 quadrature rows per point (32 points / block = 96 rows) --------
__global__ __launch_bounds__(256) void dp_mlp_kernel(
    const float* __restrict__ tx, const float* __restrict__ W0,
    const float* __restrict__ b0, const float* __restrict__ Wh,
    const float* __restrict__ bh, const float* __restrict__ Wo,
    const float* __restrict__ bo, const float* __restrict__ phi0,
    float* __restrict__ out) {
  constexpr int P = 32, MROWS = 96, NT = 8, TPW = 6, F = 6;
  __shared__ _Float16 sWT[128 * SW];     // 34816 B
  __shared__ _Float16 sH[MROWS * SH];    // 26112 B
  __shared__ float sB[128];
  __shared__ float sWo[128];
  __shared__ float sV[MROWS];
  const int tid = threadIdx.x;
  const int wave = tid >> 5, lane = tid & 31, lm = lane & 15, lg = lane >> 4;
  const int base = blockIdx.x * P;

  load_wt0<F>(sWT, W0, tid);
  if (tid < 128) { sB[tid] = b0[tid]; sWo[tid] = Wo[tid]; }
  __syncthreads();

  v8f   acc[TPW];
  float biasr[TPW];

  // ---- layer 0 ----
  {
    v16h afrag;
    int cur_mt = -1;
    #pragma unroll
    for (int i = 0; i < TPW; ++i) {
      const int tile = wave * TPW + i;
      const int mt = tile / NT, nt = tile % NT;
      if (mt != cur_mt) {                // wave-uniform rebuild when m-tile changes
        cur_mt = mt;
        #pragma unroll
        for (int h = 0; h < 16; ++h) afrag[h] = (_Float16)0.0f;
        if (lg == 0) {
          const int rl = mt * 16 + lm;   // local row = point*3 + node
          const int p = rl / 3, j = rl - 3 * p;
          const float* tp = tx + 4 * (base + p);
          const float t = tp[0];
          const float node = (j == 0) ? -0.7745966692414834f
                           : (j == 1) ? 0.0f : 0.7745966692414834f;
          float f[F];
          f[0] = 0.5f * t * (node + 1.0f);   // a*node + b with t0 = 0
          sph_features(tp, f + 1);
          #pragma unroll
          for (int k = 0; k < F; ++k) afrag[k] = (_Float16)f[k];
        }
      }
      v16h b = ld_bfrag(sWT + (nt * 16 + lm) * SW + lg * 16);
      v8f c;
      #pragma unroll
      for (int v = 0; v < 8; ++v) c[v] = 0.0f;
      acc[i]   = wmma_f16(afrag, b, c);
      biasr[i] = sB[nt * 16 + lm];
    }
  }
  __syncthreads();
  store_act<MROWS>(sH, tid, acc, biasr);
  load_wt128(sWT, Wh, tid);
  if (tid < 128) sB[tid] = bh[tid];
  __syncthreads();

  // ---- hidden layers ----
  #pragma unroll 1
  for (int l = 0; l < NSTACK; ++l) {
    gemm_hidden<MROWS>(sH, sWT, sB, tid, acc, biasr);
    __syncthreads();
    store_act<MROWS>(sH, tid, acc, biasr);
    if (l + 1 < NSTACK) {
      load_wt128(sWT, Wh + (l + 1) * 128 * 128, tid);
      if (tid < 128) sB[tid] = bh[(l + 1) * 128 + tid];
    }
    __syncthreads();
  }

  // ---- output layer + Gauss-Legendre combine ----
  if (tid < MROWS) {
    float a = bo[0];
    #pragma unroll 8
    for (int k = 0; k < 128; ++k) a += (float)sH[tid * SH + k] * sWo[k];
    sV[tid] = a;
  }
  __syncthreads();
  if (tid < P) {
    const int pt = base + tid;
    const float t = tx[4 * pt];
    const float s = (5.0f / 9.0f) * sV[3 * tid] +
                    (8.0f / 9.0f) * sV[3 * tid + 1] +
                    (5.0f / 9.0f) * sV[3 * tid + 2];
    out[pt] = phi0[pt] + 0.5f * t * s;
  }
}

extern "C" void kernel_launch(void* const* d_in, const int* in_sizes, int n_in,
                              void* d_out, int out_size, void* d_ws, size_t ws_size,
                              hipStream_t stream) {
  (void)n_in; (void)out_size; (void)ws_size;
  const float* tx    = (const float*)d_in[0];
  const float* dp_W0 = (const float*)d_in[1];
  const float* dp_b0 = (const float*)d_in[2];
  const float* dp_Wh = (const float*)d_in[3];
  const float* dp_bh = (const float*)d_in[4];
  const float* dp_Wo = (const float*)d_in[5];
  const float* dp_bo = (const float*)d_in[6];
  const float* ic_W0 = (const float*)d_in[7];
  const float* ic_b0 = (const float*)d_in[8];
  const float* ic_Wh = (const float*)d_in[9];
  const float* ic_bh = (const float*)d_in[10];
  const float* ic_Wo = (const float*)d_in[11];
  const float* ic_bo = (const float*)d_in[12];
  float* out  = (float*)d_out;
  float* phi0 = (float*)d_ws;    // N floats of scratch
  const int n = in_sizes[0] / 4; // tx_cart is (N,4)

  ic_mlp_kernel<<<n / 64, 256, 0, stream>>>(tx, ic_W0, ic_b0, ic_Wh, ic_bh,
                                            ic_Wo, ic_bo, phi0);
  dp_mlp_kernel<<<n / 32, 256, 0, stream>>>(tx, dp_W0, dp_b0, dp_Wh, dp_bh,
                                            dp_Wo, dp_bo, phi0, out);
}
